// SimpleGateLoopLayer_73418170958328
// MI455X (gfx1250) — matrix operand order repaired and checked by reading, not compiled, not run on hardware
//
#include <hip/hip_runtime.h>
#include <hip/hip_bf16.h>
#include <math.h>

// Problem constants (fixed by the reference)
constexpr int DIMD   = 1024;
constexpr int BATCH  = 2;
constexpr int SEQ    = 4096;
constexpr int MROWS  = BATCH * SEQ;   // 8192
constexpr int ECOLS  = 3 * DIMD;      // 3072
constexpr int NCHUNK = 32;
constexpr int CHUNK  = SEQ / NCHUNK;  // 128

// GEMM tiling (bf16, K-step 32)
constexpr int BM  = 64;
constexpr int BN  = 128;
constexpr int BK  = 32;
constexpr int NKB = DIMD / BK;        // 32
constexpr int AP  = 40;               // LDS pitch in halfs (80B rows -> 16B aligned chunks)
constexpr int BP  = 40;

typedef float  v8f  __attribute__((ext_vector_type(8)));
typedef __bf16 v8bf __attribute__((ext_vector_type(8)));
typedef __bf16 v16bf __attribute__((ext_vector_type(16)));
typedef int    v4i  __attribute__((vector_size(16)));   // matches builtin's pointee type

// ---- gfx1250 async global->LDS path (guarded; sync fallback keeps it compiling)
#if defined(__has_builtin)
#if __has_builtin(__builtin_amdgcn_global_load_async_to_lds_b128)
#define ASYNC_LDS 1
#endif
#endif

typedef __attribute__((address_space(1))) v4i* g_v4i_ptr;
typedef __attribute__((address_space(3))) v4i* l_v4i_ptr;

__device__ __forceinline__ void cp16(void* lds_dst, const void* g_src) {
#ifdef ASYNC_LDS
    __builtin_amdgcn_global_load_async_to_lds_b128(
        (g_v4i_ptr)g_src, (l_v4i_ptr)lds_dst, 0, 0);
#else
    *(float4*)lds_dst = *(const float4*)g_src;
#endif
}

#ifdef ASYNC_LDS
#if __has_builtin(__builtin_amdgcn_s_wait_asynccnt)
#define WAIT_ASYNC(n) __builtin_amdgcn_s_wait_asynccnt(n)
#else
#define WAIT_ASYNC(n) asm volatile("s_wait_asynccnt %0" ::"n"(n))
#endif
#else
#define WAIT_ASYNC(n)
#endif

// ---------------- Kernel 1: RMSNorm + split xn into bf16 hi/lo planes -------
__global__ __launch_bounds__(256)
void xn_split_kernel(const float* __restrict__ x, const float* __restrict__ gamma,
                     __bf16* __restrict__ xh, __bf16* __restrict__ xl) {
    const int row  = blockIdx.x;
    const int tid  = threadIdx.x;
    const int lane = tid & 31;
    const int wid  = tid >> 5;
    const float* r = x + (size_t)row * DIMD;

    float v[4];
    float sum = 0.f;
    #pragma unroll
    for (int i = 0; i < 4; ++i) {
        v[i] = r[tid + i * 256];
        sum = fmaf(v[i], v[i], sum);
    }
    #pragma unroll
    for (int off = 16; off > 0; off >>= 1) sum += __shfl_xor(sum, off, 32);

    __shared__ float red[8];
    if (lane == 0) red[wid] = sum;
    __syncthreads();
    float tot = 0.f;
    #pragma unroll
    for (int wv = 0; wv < 8; ++wv) tot += red[wv];

    const float s = 32.0f / fmaxf(sqrtf(tot), 1e-12f);   // sqrt(1024)=32
    #pragma unroll
    for (int i = 0; i < 4; ++i) {
        const int d = tid + i * 256;
        const float xn = v[i] * s * gamma[d];
        const __bf16 hi = (__bf16)xn;
        const __bf16 lo = (__bf16)(xn - (float)hi);
        xh[(size_t)row * DIMD + d] = hi;
        xl[(size_t)row * DIMD + d] = lo;
    }
}

// ---------------- Kernel 2: split w into bf16 hi/lo planes -------------------
__global__ __launch_bounds__(256)
void w_split_kernel(const float* __restrict__ w, __bf16* __restrict__ wh,
                    __bf16* __restrict__ wl) {
    const size_t i = (size_t)blockIdx.x * 256 + threadIdx.x;
    const float v = w[i];
    const __bf16 hi = (__bf16)v;
    const __bf16 lo = (__bf16)(v - (float)hi);
    wh[i] = hi;
    wl[i] = lo;
}

// ---------------- Fragment loaders (bf16 16x16x32 A/B layouts, wave32) -------
__device__ __forceinline__ v16bf ldfragA(const __bf16* row, int h) {
    // lane(j,h): elems 0..7 = K = h*8..h*8+7 ; elems 8..15 = K = 16+h*8..16+h*8+7
    const v8bf lo = *(const v8bf*)(row + h * 8);
    const v8bf hi = *(const v8bf*)(row + 16 + h * 8);
    return __builtin_shufflevector(lo, hi, 0, 1, 2, 3, 4, 5, 6, 7,
                                           8, 9, 10, 11, 12, 13, 14, 15);
}
__device__ __forceinline__ v16bf ldfragB(const __bf16* row, int h) {
    // lane(j,h): elems 0..7 = K = h*8..h*8+7 ; elems 8..15 = K = 16+h*8..16+h*8+7, N = j  (row = LDS row of column N)
    const v8bf lo = *(const v8bf*)(row + h * 8);
    const v8bf hi = *(const v8bf*)(row + 16 + h * 8);
    return __builtin_shufflevector(lo, hi, 0, 1, 2, 3, 4, 5, 6, 7,
                                           8, 9, 10, 11, 12, 13, 14, 15);
}

// ---------------- Kernel 3: bf16x3 WMMA GEMM (double-buffered async LDS) -----
__global__ __launch_bounds__(256)
void qkva_gemm_kernel(const __bf16* __restrict__ xh, const __bf16* __restrict__ xl,
                      const __bf16* __restrict__ wh, const __bf16* __restrict__ wl,
                      float* __restrict__ qout, float* __restrict__ kvout,
                      float* __restrict__ aout) {
    __shared__ __align__(16) __bf16 AH[2][BM * AP];
    __shared__ __align__(16) __bf16 AL[2][BM * AP];
    __shared__ __align__(16) __bf16 BH[2][BN * BP];
    __shared__ __align__(16) __bf16 BL[2][BN * BP];

    const int tid   = threadIdx.x;
    const int wid   = tid >> 5;
    const int lane  = tid & 31;
    const int j     = lane & 15;
    const int h     = lane >> 4;
    const int waveM = wid & 3;          // 4 waves along M
    const int waveN = wid >> 2;         // 2 waves along N

    const int m0 = blockIdx.x * BM;
    const int e0 = blockIdx.y * BN;

    // Staging coords: 16B (8-half) chunks
    const int am = tid >> 2;            // 0..63
    const int ak = (tid & 3) * 8;       // 0,8,16,24

    auto stage = [&](int buf, int k0) {
        {   // A planes: 64x32 halfs each, 1 chunk/thread/plane
            const size_t g = (size_t)(m0 + am) * DIMD + k0 + ak;
            cp16(&AH[buf][am * AP + ak], xh + g);
            cp16(&AL[buf][am * AP + ak], xl + g);
        }
        #pragma unroll
        for (int c = 0; c < 2; ++c) {   // B planes: 128x32 halfs each, 2 chunks/thread/plane
            const int id = tid + c * 256;
            const int e  = id >> 2;     // 0..127
            const int kc = (id & 3) * 8;
            const size_t g = (size_t)(e0 + e) * DIMD + k0 + kc;
            cp16(&BH[buf][e * BP + kc], wh + g);
            cp16(&BL[buf][e * BP + kc], wl + g);
        }
    };  // 6 async instructions per wave per stage

    v8f acc[4] = {};

    stage(0, 0);
    for (int kb = 0; kb < NKB; ++kb) {
        const int cur = kb & 1;
        const bool hasNext = (kb + 1) < NKB;
        if (hasNext) {
            stage(cur ^ 1, (kb + 1) * BK);
            WAIT_ASYNC(6);              // retire current buffer's 6 copies
        } else {
            WAIT_ASYNC(0);
        }
        __syncthreads();

        const v16bf aHf = ldfragA(&AH[cur][(waveM * 16 + j) * AP], h);
        const v16bf aLf = ldfragA(&AL[cur][(waveM * 16 + j) * AP], h);
        #pragma unroll
        for (int t = 0; t < 4; ++t) {
            const int n = waveN * 64 + t * 16 + j;
            const v16bf bHf = ldfragB(&BH[cur][n * BP], h);
            const v16bf bLf = ldfragB(&BL[cur][n * BP], h);
            // fp32-accurate product: hi*hi + lo*hi + hi*lo (lo*lo ~ 2^-32, dropped)
            acc[t] = __builtin_amdgcn_wmma_f32_16x16x32_bf16(
                false, aHf, false, bHf, (short)0, acc[t], false, false);
            acc[t] = __builtin_amdgcn_wmma_f32_16x16x32_bf16(
                false, aLf, false, bHf, (short)0, acc[t], false, false);
            acc[t] = __builtin_amdgcn_wmma_f32_16x16x32_bf16(
                false, aHf, false, bLf, (short)0, acc[t], false, false);
        }
        __syncthreads();
    }

    // Epilogue: 128-col block tiles align exactly with q/kv/a sections
    const int sec     = blockIdx.y >> 3;                 // 0=q, 1=kv, 2=a
    const int colsec0 = (blockIdx.y & 7) * BN + waveN * 64;
    float* dstbase = (sec == 0) ? qout : (sec == 1) ? kvout : aout;
    #pragma unroll
    for (int t = 0; t < 4; ++t) {
        const int col = colsec0 + t * 16 + j;
        #pragma unroll
        for (int r = 0; r < 8; ++r) {
            const int mg = m0 + waveM * 16 + 8 * h + r;
            float v = acc[t][r];
            if (sec == 2) v = 1.0f / (1.0f + __expf(-v));  // fused sigmoid
            dstbase[(size_t)mg * DIMD + col] = v;
        }
    }
}

// ---------------- Kernel 4: per-chunk (A, KV) aggregates ---------------------
__global__ __launch_bounds__(256)
void chunk_reduce_kernel(const float* __restrict__ wa, const float* __restrict__ wkv,
                         float* __restrict__ cA, float* __restrict__ cKV) {
    const int d = blockIdx.x * blockDim.x + threadIdx.x;
    const int c = blockIdx.y;
    const int b = blockIdx.z;
    size_t idx = ((size_t)(b * SEQ + c * CHUNK)) * DIMD + d;
    float A = 1.f, KV = 0.f;
    for (int i = 0; i < CHUNK; ++i) {
        const float a  = wa[idx];
        const float kv = wkv[idx];
        KV = fmaf(a, KV, kv);
        A *= a;
        idx += DIMD;
    }
    const size_t o = ((size_t)(b * NCHUNK + c)) * DIMD + d;
    cA[o] = A;
    cKV[o] = KV;
}

// ---------------- Kernel 5: exclusive scan over chunk aggregates -------------
__global__ __launch_bounds__(256)
void chunk_scan_kernel(const float* __restrict__ cA, const float* __restrict__ cKV,
                       float* __restrict__ pKV) {
    const int t = blockIdx.x * blockDim.x + threadIdx.x;   // 0..2047
    const int b = t / DIMD, d = t % DIMD;
    float carry = 0.f;
    for (int c = 0; c < NCHUNK; ++c) {
        const size_t o = ((size_t)(b * NCHUNK + c)) * DIMD + d;
        pKV[o] = carry;
        carry = fmaf(cA[o], carry, cKV[o]);
    }
}

// ---------------- Kernel 6: rebuild kv_scan per chunk, out = q * kv_scan -----
__global__ __launch_bounds__(256)
void apply_kernel(const float* __restrict__ wa, const float* __restrict__ wkv,
                  const float* __restrict__ pKV, float* __restrict__ out) {
    const int d = blockIdx.x * blockDim.x + threadIdx.x;
    const int c = blockIdx.y;
    const int b = blockIdx.z;
    float carry = pKV[((size_t)(b * NCHUNK + c)) * DIMD + d];
    size_t idx = ((size_t)(b * SEQ + c * CHUNK)) * DIMD + d;
    for (int i = 0; i < CHUNK; ++i) {
        carry = fmaf(wa[idx], carry, wkv[idx]);
        out[idx] *= carry;                                  // out held q
        idx += DIMD;
    }
}

extern "C" void kernel_launch(void* const* d_in, const int* in_sizes, int n_in,
                              void* d_out, int out_size, void* d_ws, size_t ws_size,
                              hipStream_t stream) {
    const float* x     = (const float*)d_in[0];   // (2,4096,1024)
    const float* gamma = (const float*)d_in[1];   // (1024,)
    const float* w     = (const float*)d_in[2];   // (3072,1024)
    float* out = (float*)d_out;

    // Deterministic workspace carve (256B aligned)
    char* p = (char*)d_ws;
    auto carve = [&](size_t bytes) {
        void* r = (void*)p;
        p += (bytes + 255) & ~(size_t)255;
        return r;
    };
    __bf16* xh  = (__bf16*)carve((size_t)MROWS * DIMD * 2);
    __bf16* xl  = (__bf16*)carve((size_t)MROWS * DIMD * 2);
    __bf16* wh  = (__bf16*)carve((size_t)ECOLS * DIMD * 2);
    __bf16* wl  = (__bf16*)carve((size_t)ECOLS * DIMD * 2);
    float*  kv  = (float*)carve((size_t)MROWS * DIMD * 4);
    float*  a   = (float*)carve((size_t)MROWS * DIMD * 4);
    float*  cA  = (float*)carve((size_t)BATCH * NCHUNK * DIMD * 4);
    float*  cKV = (float*)carve((size_t)BATCH * NCHUNK * DIMD * 4);
    float*  pKV = (float*)carve((size_t)BATCH * NCHUNK * DIMD * 4);

    xn_split_kernel<<<MROWS, 256, 0, stream>>>(x, gamma, xh, xl);
    w_split_kernel<<<(ECOLS * DIMD) / 256, 256, 0, stream>>>(w, wh, wl);

    dim3 ggrid(MROWS / BM, ECOLS / BN);           // 128 x 24
    qkva_gemm_kernel<<<ggrid, 256, 0, stream>>>(xh, xl, wh, wl, out, kv, a);

    chunk_reduce_kernel<<<dim3(DIMD / 256, NCHUNK, BATCH), 256, 0, stream>>>(a, kv, cA, cKV);
    chunk_scan_kernel<<<(BATCH * DIMD) / 256, 256, 0, stream>>>(cA, cKV, pKV);
    apply_kernel<<<dim3(DIMD / 256, NCHUNK, BATCH), 256, 0, stream>>>(a, kv, pKV, out);
}
